// KANGPS_Hybrid_layer_62989990363554
// MI455X (gfx1250) — matrix operand order repaired and checked by reading, not compiled, not run on hardware
//
#include <hip/hip_runtime.h>
#include <hip/hip_bf16.h>
#include <math.h>

#define N_NODES 4096
#define DFEAT   256
#define NEDGE   131072
#define NHEAD   8
#define HIDF    512
#define NCOEF   7
#define KSPL    (DFEAT * NCOEF)   /* 1792 */

typedef __bf16 v16bf __attribute__((ext_vector_type(16)));
typedef float  v8f   __attribute__((ext_vector_type(8)));
typedef int    v4i   __attribute__((vector_size(16)));

#if __has_builtin(__builtin_amdgcn_global_load_async_to_lds_b128) && \
    __has_builtin(__builtin_amdgcn_s_wait_asynccnt)
#define HAVE_ASYNC_LDS 1
typedef __attribute__((address_space(1))) v4i* gvec_p;
typedef __attribute__((address_space(3))) v4i* lvec_p;
#else
#define HAVE_ASYNC_LDS 0
#endif

__device__ __forceinline__ float silu_f(float v) { return v / (1.0f + __expf(-v)); }

// ---- DPP ROW_XMASK butterfly reductions over 16-lane groups (no LDS) ----
template <int MASK>
__device__ __forceinline__ float xmask_f(float x) {
    int r = __builtin_amdgcn_update_dpp(0, __float_as_int(x),
                                        0x160 | MASK, 0xf, 0xf, true);
    return __int_as_float(r);
}
__device__ __forceinline__ float red_max16(float x) {
    x = fmaxf(x, xmask_f<1>(x));
    x = fmaxf(x, xmask_f<2>(x));
    x = fmaxf(x, xmask_f<4>(x));
    x = fmaxf(x, xmask_f<8>(x));
    return x;
}
__device__ __forceinline__ float red_sum16(float x) {
    x += xmask_f<1>(x);
    x += xmask_f<2>(x);
    x += xmask_f<4>(x);
    x += xmask_f<8>(x);
    return x;
}

// ---- WMMA fragment loaders (wave32, CDNA5 ISA 7.12.2) ----
__device__ __forceinline__ v16bf load_frag_a(const __bf16* As, int lda) {
    int lane = threadIdx.x & 31;
    int r  = lane & 15;
    int kb = (lane >> 4) << 3;
    const __bf16* p = As + r * lda;
    v16bf a;
#pragma unroll
    for (int i = 0; i < 8; ++i) a[i]     = p[kb + i];
#pragma unroll
    for (int i = 0; i < 8; ++i) a[8 + i] = p[16 + kb + i];
    return a;
}
__device__ __forceinline__ v16bf load_frag_b(const __bf16* Bs, int ldb, int ncol) {
    int lane = threadIdx.x & 31;
    int kb = (lane >> 4) << 3;
    const __bf16* p = Bs + ncol + (lane & 15);
    v16bf b;
#pragma unroll
    for (int i = 0; i < 8; ++i) b[i]     = p[(kb + i) * ldb];
#pragma unroll
    for (int i = 0; i < 8; ++i) b[8 + i] = p[(16 + kb + i) * ldb];
    return b;
}

// =====================================================================
// WMMA GEMM: out[M,Nt] = epi( A[M,Kt] @ Wt[Kt,Nt] ), Wt pre-transposed bf16.
// Block tile 64x64, 8 waves (4x2), 2 accumulators per wave.
// =====================================================================
template <typename TA, bool PRE_SILU, bool POST_SILU, bool ACCUM>
__global__ __launch_bounds__(256)
void gemm_wmma(const TA* __restrict__ A, const __bf16* __restrict__ Wt,
               const float* __restrict__ bias,
               const float* __restrict__ bn_g, const float* __restrict__ bn_b,
               const float* __restrict__ res,
               float* __restrict__ out, __bf16* __restrict__ out_bf,
               int M, int Nt, int Kt)
{
    __shared__ __bf16 As[64][32];
    __shared__ __bf16 Bs[32][64];
    const int m0  = blockIdx.y * 64;
    const int n0  = blockIdx.x * 64;
    const int tid = threadIdx.x;
    const int wave = tid >> 5;
    const int wm = wave >> 1, wn = wave & 1;

    v8f acc0 = {}, acc1 = {};
    for (int k0 = 0; k0 < Kt; k0 += 32) {
        // ---- stage B tile (32x64 bf16): contiguous 16B per thread ----
        {
            int k   = tid >> 3;
            int seg = tid & 7;
            const __bf16* src = Wt + (size_t)(k0 + k) * Nt + n0 + seg * 8;
#if HAVE_ASYNC_LDS
            __builtin_amdgcn_global_load_async_to_lds_b128(
                (gvec_p)(void*)(const void*)src, (lvec_p)&Bs[k][seg * 8], 0, 0);
#else
            *(uint4*)&Bs[k][seg * 8] = *(const uint4*)src;
#endif
        }
        // ---- stage A tile (64x32) ----
        if constexpr (sizeof(TA) == 2) {           // bf16 source: raw 16B copies
            int m   = tid >> 2;
            int seg = tid & 3;
            const TA* src = A + (size_t)(m0 + m) * Kt + k0 + seg * 8;
#if HAVE_ASYNC_LDS
            __builtin_amdgcn_global_load_async_to_lds_b128(
                (gvec_p)(void*)(const void*)src, (lvec_p)&As[m][seg * 8], 0, 0);
#else
            *(uint4*)&As[m][seg * 8] = *(const uint4*)src;
#endif
        } else {                                   // f32 source: convert (+SiLU)
            int m  = tid >> 2;
            int kk = (tid & 3) << 3;
            const TA* src = A + (size_t)(m0 + m) * Kt + k0 + kk;
            if (k0 + 32 < Kt) __builtin_prefetch(src + 32, 0, 1);
            alignas(16) __bf16 tmp[8];
#pragma unroll
            for (int j = 0; j < 8; ++j) {
                float v = (float)src[j];
                if (PRE_SILU) v = silu_f(v);
                tmp[j] = (__bf16)v;
            }
            *(uint4*)&As[m][kk] = *(const uint4*)tmp;
        }
#if HAVE_ASYNC_LDS
        __builtin_amdgcn_s_wait_asynccnt(0);
#endif
        __syncthreads();

        v16bf a  = load_frag_a(&As[wm * 16][0], 32);
        v16bf b0 = load_frag_b(&Bs[0][0], 64, wn * 32);
        v16bf b1 = load_frag_b(&Bs[0][0], 64, wn * 32 + 16);
        acc0 = __builtin_amdgcn_wmma_f32_16x16x32_bf16(false, a, false, b0,
                                                       (short)0, acc0, false, false);
        acc1 = __builtin_amdgcn_wmma_f32_16x16x32_bf16(false, a, false, b1,
                                                       (short)0, acc1, false, false);
        __syncthreads();
    }

    // ---- fused epilogue ----
    const int lane = tid & 31;
    const int half = lane >> 4;
    const int c0   = n0 + wn * 32 + (lane & 15);
    const int c1   = c0 + 16;
    const float bnscale = rsqrtf(1.0f + 1e-5f);
#pragma unroll
    for (int v = 0; v < 8; ++v) {
        int row = m0 + wm * 16 + v + half * 8;
        float v0 = acc0[v], v1 = acc1[v];
        if (bias) { v0 += bias[c0]; v1 += bias[c1]; }
        if (POST_SILU) { v0 = silu_f(v0); v1 = silu_f(v1); }
        if (bn_g) {
            v0 = v0 * bn_g[c0] * bnscale + bn_b[c0];
            v1 = v1 * bn_g[c1] * bnscale + bn_b[c1];
        }
        size_t i0 = (size_t)row * Nt + c0;
        size_t i1 = (size_t)row * Nt + c1;
        if (res)   { v0 += res[i0]; v1 += res[i1]; }
        if (ACCUM) { v0 += out[i0]; v1 += out[i1]; }
        out[i0] = v0; out[i1] = v1;
        if (out_bf) { out_bf[i0] = (__bf16)v0; out_bf[i1] = (__bf16)v1; }
    }
}

// =====================================================================
// Flash attention: one wave per (head, 16-query tile). dh=32, bf16 qkv.
// =====================================================================
__global__ __launch_bounds__(32)
void flash_attn_k(const __bf16* __restrict__ qkvb, float* __restrict__ o)
{
    __shared__ __bf16 Qs[16][32];
    __shared__ __bf16 Kst[32][32];  // [dim][key] -> B frag for Q@K^T
    __shared__ __bf16 Vs[32][32];   // [key][dim] -> B frag for P@V
    __shared__ __bf16 Ps[16][32];

    const int h    = blockIdx.y;
    const int q0   = blockIdx.x * 16;
    const int lane = threadIdx.x;
    const float sc = 0.17677669529663687f;  // dh^-0.5

    {   // stage Q (scaled), packed 16B stores
        int r  = lane & 15;
        int d0 = (lane >> 4) * 16;
        const __bf16* src = qkvb + (size_t)(q0 + r) * 768 + h * 32 + d0;
        alignas(16) __bf16 tmp[16];
#pragma unroll
        for (int j = 0; j < 16; ++j) tmp[j] = (__bf16)((float)src[j] * sc);
        *(uint4*)&Qs[r][d0]     = *(const uint4*)&tmp[0];
        *(uint4*)&Qs[r][d0 + 8] = *(const uint4*)&tmp[8];
    }
    __syncthreads();
    v16bf aq = load_frag_a(&Qs[0][0], 32);

    v8f o0 = {}, o1 = {};
    float mrow[8], lrow[8];
#pragma unroll
    for (int v = 0; v < 8; ++v) { mrow[v] = -1e30f; lrow[v] = 0.0f; }

    for (int kc = 0; kc < N_NODES; kc += 32) {
        __syncthreads();
        {   // lane = key index: V rows contiguous; K transposed via register scatter
            const __bf16* vp = qkvb + (size_t)(kc + lane) * 768 + 512 + h * 32;
#if HAVE_ASYNC_LDS
#pragma unroll
            for (int s = 0; s < 4; ++s)
                __builtin_amdgcn_global_load_async_to_lds_b128(
                    (gvec_p)(void*)(const void*)(vp + s * 8), (lvec_p)&Vs[lane][s * 8], 0, 0);
#else
#pragma unroll
            for (int s = 0; s < 4; ++s)
                *(uint4*)&Vs[lane][s * 8] = *(const uint4*)(vp + s * 8);
#endif
            const __bf16* kp = qkvb + (size_t)(kc + lane) * 768 + 256 + h * 32;
            alignas(16) __bf16 kr[32];
#pragma unroll
            for (int s = 0; s < 4; ++s)
                *(uint4*)&kr[s * 8] = *(const uint4*)(kp + s * 8);
#pragma unroll
            for (int d = 0; d < 32; ++d) Kst[d][lane] = kr[d];
        }
#if HAVE_ASYNC_LDS
        __builtin_amdgcn_s_wait_asynccnt(0);
#endif
        __syncthreads();

        v16bf bk0 = load_frag_b(&Kst[0][0], 32, 0);
        v16bf bk1 = load_frag_b(&Kst[0][0], 32, 16);
        v8f zero = {};
        v8f s0 = __builtin_amdgcn_wmma_f32_16x16x32_bf16(false, aq, false, bk0,
                                                         (short)0, zero, false, false);
        v8f s1 = __builtin_amdgcn_wmma_f32_16x16x32_bf16(false, aq, false, bk1,
                                                         (short)0, zero, false, false);
        // online softmax via DPP xmask reductions (rows live in 16-lane halves)
        float p0[8], p1[8];
#pragma unroll
        for (int v = 0; v < 8; ++v) {
            float mx = red_max16(fmaxf(s0[v], s1[v]));
            float mnew  = fmaxf(mrow[v], mx);
            float scale = __expf(mrow[v] - mnew);
            p0[v] = __expf(s0[v] - mnew);
            p1[v] = __expf(s1[v] - mnew);
            float rs = red_sum16(p0[v] + p1[v]);
            lrow[v] = lrow[v] * scale + rs;
            mrow[v] = mnew;
            o0[v] *= scale; o1[v] *= scale;
        }
        {   // spill P (C-layout) -> row-major LDS for A-frag reload
            int hh = lane >> 4;
            int c  = lane & 15;
#pragma unroll
            for (int v = 0; v < 8; ++v) {
                Ps[v + hh * 8][c]      = (__bf16)p0[v];
                Ps[v + hh * 8][16 + c] = (__bf16)p1[v];
            }
        }
        __syncthreads();
        v16bf ap  = load_frag_a(&Ps[0][0], 32);
        v16bf bv0 = load_frag_b(&Vs[0][0], 32, 0);
        v16bf bv1 = load_frag_b(&Vs[0][0], 32, 16);
        o0 = __builtin_amdgcn_wmma_f32_16x16x32_bf16(false, ap, false, bv0,
                                                     (short)0, o0, false, false);
        o1 = __builtin_amdgcn_wmma_f32_16x16x32_bf16(false, ap, false, bv1,
                                                     (short)0, o1, false, false);
    }

    const int hh = lane >> 4;
    const int c  = lane & 15;
#pragma unroll
    for (int v = 0; v < 8; ++v) {
        int row = q0 + v + hh * 8;
        float inv = 1.0f / lrow[v];
        o[(size_t)row * DFEAT + h * 32 + c]      = o0[v] * inv;
        o[(size_t)row * DFEAT + h * 32 + 16 + c] = o1[v] * inv;
    }
}

// =====================================================================
// Prep / graph / elementwise kernels
// =====================================================================
// W[Nt][Kt] f32 -> Wt[Kt][Nt] bf16
__global__ void cvt_transpose_k(const float* __restrict__ W, __bf16* __restrict__ Wt,
                                int Nt, int Kt) {
    int idx = blockIdx.x * 256 + threadIdx.x;
    if (idx >= Nt * Kt) return;
    int n = idx / Kt, k = idx % Kt;
    Wt[(size_t)k * Nt + n] = (__bf16)W[idx];
}
// (spline*scaler)[o][i][c] -> Wt[i*7+c][o] bf16
__global__ void flatten_spline_t_k(const float* __restrict__ sp, const float* __restrict__ scl,
                                   __bf16* __restrict__ Wt) {
    int idx = blockIdx.x * 256 + threadIdx.x;
    if (idx >= DFEAT * KSPL) return;
    int o = idx / KSPL, r = idx % KSPL;
    Wt[(size_t)r * DFEAT + o] = (__bf16)(sp[idx] * scl[o * DFEAT + r / NCOEF]);
}
__global__ void deg_init_k(float* deg) {
    int i = blockIdx.x * 256 + threadIdx.x;
    if (i < N_NODES) deg[i] = 1.0f;
}
__global__ void deg_count_k(const int* __restrict__ ei, float* deg) {
    int e = blockIdx.x * 256 + threadIdx.x;
    if (e < NEDGE) atomicAdd(&deg[ei[e]], 1.0f);
}
__global__ void dis_k(const float* __restrict__ deg, float* dis) {
    int i = blockIdx.x * 256 + threadIdx.x;
    if (i < N_NODES) dis[i] = rsqrtf(deg[i]);
}
__global__ __launch_bounds__(256)
void self_loop_k(const float* __restrict__ x, const float* __restrict__ dis,
                 float* __restrict__ aggr) {
    int i = blockIdx.x, d = threadIdx.x;
    float c = dis[i] * dis[i];
    aggr[(size_t)i * DFEAT + d] = c * x[(size_t)i * DFEAT + d];
}
__global__ __launch_bounds__(256)
void edge_aggr_k(const float* __restrict__ x, const int* __restrict__ ei,
                 const float* __restrict__ dis, float* __restrict__ aggr) {
    int e = blockIdx.x, d = threadIdx.x;
    int s = ei[e], t = ei[NEDGE + e];
    float c = dis[s] * dis[t];
    atomicAdd(&aggr[(size_t)t * DFEAT + d], c * x[(size_t)s * DFEAT + d]);
}
__global__ __launch_bounds__(256)
void layernorm_k(const float* __restrict__ x, const float* __restrict__ g,
                 const float* __restrict__ b, float* __restrict__ out) {
    __shared__ float red[8];
    int row = blockIdx.x, tid = threadIdx.x;
    float v = x[(size_t)row * DFEAT + tid];
    float s = v;
#pragma unroll
    for (int off = 16; off >= 1; off >>= 1) s += __shfl_xor(s, off, 32);
    if ((tid & 31) == 0) red[tid >> 5] = s;
    __syncthreads();
    float tot = 0.0f;
#pragma unroll
    for (int i = 0; i < 8; ++i) tot += red[i];
    float mu = tot * (1.0f / DFEAT);
    __syncthreads();
    float d = v - mu;
    float s2 = d * d;
#pragma unroll
    for (int off = 16; off >= 1; off >>= 1) s2 += __shfl_xor(s2, off, 32);
    if ((tid & 31) == 0) red[tid >> 5] = s2;
    __syncthreads();
    float tv = 0.0f;
#pragma unroll
    for (int i = 0; i < 8; ++i) tv += red[i];
    out[(size_t)row * DFEAT + tid] = d * rsqrtf(tv * (1.0f / DFEAT) + 1e-5f) * g[tid] + b[tid];
}
__global__ void add_k(const float* __restrict__ a, const float* __restrict__ b,
                      float* __restrict__ o, int n) {
    int i = blockIdx.x * 256 + threadIdx.x;
    if (i < n) o[i] = a[i] + b[i];
}
// Cox-de Boor order-3 uniform B-spline bases -> bf16 [n*D][7]
__global__ void bspline_k(const float* __restrict__ h, __bf16* __restrict__ bs, int total) {
    int idx = blockIdx.x * 256 + threadIdx.x;
    if (idx >= total) return;
    float x = h[idx];
    const float g0 = -2.5f;
    float b[10];
#pragma unroll
    for (int t = 0; t < 10; ++t) {
        float gt = g0 + 0.5f * t;
        b[t] = (x >= gt && x < gt + 0.5f) ? 1.0f : 0.0f;
    }
#pragma unroll
    for (int kk = 1; kk <= 3; ++kk) {
        float inv = 1.0f / (0.5f * kk);
#pragma unroll
        for (int t = 0; t < 10 - kk; ++t) {
            float gt = g0 + 0.5f * t;
            b[t] = (x - gt) * inv * b[t] + (gt + 0.5f * (kk + 1) - x) * inv * b[t + 1];
        }
    }
#pragma unroll
    for (int c = 0; c < NCOEF; ++c) bs[(size_t)idx * NCOEF + c] = (__bf16)b[c];
}

// =====================================================================
extern "C" void kernel_launch(void* const* d_in, const int* in_sizes, int n_in,
                              void* d_out, int out_size, void* d_ws, size_t ws_size,
                              hipStream_t stream)
{
    const float* x      = (const float*)d_in[0];
    const int*   ei     = (const int*)d_in[1];
    const float* gcn_w1 = (const float*)d_in[2];
    const float* gcn_b1 = (const float*)d_in[3];
    const float* gcn_w2 = (const float*)d_in[4];
    const float* gcn_b2 = (const float*)d_in[5];
    const float* ln_g   = (const float*)d_in[6];
    const float* ln_b   = (const float*)d_in[7];
    const float* w_qkv  = (const float*)d_in[8];
    const float* b_qkv  = (const float*)d_in[9];
    const float* w_o    = (const float*)d_in[10];
    const float* b_o    = (const float*)d_in[11];
    const float* a_w1   = (const float*)d_in[12];
    const float* a_b1   = (const float*)d_in[13];
    const float* a_w2   = (const float*)d_in[14];
    const float* a_b2   = (const float*)d_in[15];
    const float* bn1_g  = (const float*)d_in[16];
    const float* bn1_b  = (const float*)d_in[17];
    const float* bn2_g  = (const float*)d_in[18];
    const float* bn2_b  = (const float*)d_in[19];
    const float* k1_base = (const float*)d_in[20];
    const float* k1_sp   = (const float*)d_in[21];
    const float* k1_sc   = (const float*)d_in[22];
    const float* k2_base = (const float*)d_in[23];
    const float* k2_sp   = (const float*)d_in[24];
    const float* k2_sc   = (const float*)d_in[25];
    (void)in_sizes; (void)n_in; (void)out_size; (void)ws_size;

    char* ws = (char*)d_ws;
    size_t off = 0;
    auto alloc = [&](size_t bytes) -> char* {
        char* p = ws + off;
        off += (bytes + 255) & ~(size_t)255;
        return p;
    };
    float* deg  = (float*)alloc((size_t)N_NODES * 4);
    float* dis  = (float*)alloc((size_t)N_NODES * 4);
    float* aggr = (float*)alloc((size_t)N_NODES * DFEAT * 4);
    float* h1   = (float*)alloc((size_t)N_NODES * HIDF * 4);
    float* x_m  = (float*)alloc((size_t)N_NODES * DFEAT * 4);
    float* xn   = (float*)alloc((size_t)N_NODES * DFEAT * 4);
    float* qkv  = (float*)alloc((size_t)N_NODES * 3 * DFEAT * 4);
    float* ob   = (float*)alloc((size_t)N_NODES * DFEAT * 4);
    float* o2   = (float*)alloc((size_t)N_NODES * DFEAT * 4);
    float* x_t  = (float*)alloc((size_t)N_NODES * DFEAT * 4);
    float* hbuf = (float*)alloc((size_t)N_NODES * DFEAT * 4);
    float* kmid = (float*)alloc((size_t)N_NODES * DFEAT * 4);
    __bf16* qkvb = (__bf16*)alloc((size_t)N_NODES * 3 * DFEAT * 2);
    __bf16* bs   = (__bf16*)alloc((size_t)N_NODES * KSPL * 2);
    __bf16* wt_g1 = (__bf16*)alloc((size_t)HIDF * DFEAT * 2);
    __bf16* wt_g2 = (__bf16*)alloc((size_t)DFEAT * HIDF * 2);
    __bf16* wt_qkv = (__bf16*)alloc((size_t)3 * DFEAT * DFEAT * 2);
    __bf16* wt_o  = (__bf16*)alloc((size_t)DFEAT * DFEAT * 2);
    __bf16* wt_a1 = (__bf16*)alloc((size_t)HIDF * DFEAT * 2);
    __bf16* wt_a2 = (__bf16*)alloc((size_t)DFEAT * HIDF * 2);
    __bf16* wt_k1 = (__bf16*)alloc((size_t)DFEAT * DFEAT * 2);
    __bf16* wt_k2 = (__bf16*)alloc((size_t)DFEAT * DFEAT * 2);
    __bf16* wt_sp = (__bf16*)alloc((size_t)KSPL * DFEAT * 2);
    float* out = (float*)d_out;

    dim3 blk256(256);
    auto ggrid = [](int M, int Nt) { return dim3(Nt / 64, M / 64); };
    auto cgrid = [](int n) { return dim3((n + 255) / 256); };

    // ---- weight prep: f32 [Nt][Kt] -> bf16 [Kt][Nt] ----
    cvt_transpose_k<<<cgrid(HIDF * DFEAT), blk256, 0, stream>>>(gcn_w1, wt_g1, HIDF, DFEAT);
    cvt_transpose_k<<<cgrid(DFEAT * HIDF), blk256, 0, stream>>>(gcn_w2, wt_g2, DFEAT, HIDF);
    cvt_transpose_k<<<cgrid(3 * DFEAT * DFEAT), blk256, 0, stream>>>(w_qkv, wt_qkv, 3 * DFEAT, DFEAT);
    cvt_transpose_k<<<cgrid(DFEAT * DFEAT), blk256, 0, stream>>>(w_o, wt_o, DFEAT, DFEAT);
    cvt_transpose_k<<<cgrid(HIDF * DFEAT), blk256, 0, stream>>>(a_w1, wt_a1, HIDF, DFEAT);
    cvt_transpose_k<<<cgrid(DFEAT * HIDF), blk256, 0, stream>>>(a_w2, wt_a2, DFEAT, HIDF);
    cvt_transpose_k<<<cgrid(DFEAT * DFEAT), blk256, 0, stream>>>(k1_base, wt_k1, DFEAT, DFEAT);
    cvt_transpose_k<<<cgrid(DFEAT * DFEAT), blk256, 0, stream>>>(k2_base, wt_k2, DFEAT, DFEAT);

    // ---- GCN branch ----
    deg_init_k<<<N_NODES / 256, blk256, 0, stream>>>(deg);
    deg_count_k<<<NEDGE / 256, blk256, 0, stream>>>(ei, deg);
    dis_k<<<N_NODES / 256, blk256, 0, stream>>>(deg, dis);
    self_loop_k<<<N_NODES, blk256, 0, stream>>>(x, dis, aggr);
    edge_aggr_k<<<NEDGE, blk256, 0, stream>>>(x, ei, dis, aggr);
    gemm_wmma<float, false, true, false><<<ggrid(N_NODES, HIDF), blk256, 0, stream>>>(
        aggr, wt_g1, gcn_b1, nullptr, nullptr, nullptr, h1, nullptr, N_NODES, HIDF, DFEAT);
    gemm_wmma<float, false, false, false><<<ggrid(N_NODES, DFEAT), blk256, 0, stream>>>(
        h1, wt_g2, gcn_b2, bn1_g, bn1_b, x, x_m, nullptr, N_NODES, DFEAT, HIDF);

    // ---- Graphormer attention branch ----
    layernorm_k<<<N_NODES, blk256, 0, stream>>>(x, ln_g, ln_b, xn);
    gemm_wmma<float, false, false, false><<<ggrid(N_NODES, 3 * DFEAT), blk256, 0, stream>>>(
        xn, wt_qkv, b_qkv, nullptr, nullptr, nullptr, qkv, qkvb, N_NODES, 3 * DFEAT, DFEAT);
    flash_attn_k<<<dim3(N_NODES / 16, NHEAD), dim3(32), 0, stream>>>(qkvb, ob);
    gemm_wmma<float, false, false, false><<<ggrid(N_NODES, DFEAT), blk256, 0, stream>>>(
        ob, wt_o, b_o, nullptr, nullptr, nullptr, o2, nullptr, N_NODES, DFEAT, DFEAT);
    gemm_wmma<float, false, true, false><<<ggrid(N_NODES, HIDF), blk256, 0, stream>>>(
        o2, wt_a1, a_b1, nullptr, nullptr, nullptr, h1, nullptr, N_NODES, HIDF, DFEAT);
    gemm_wmma<float, false, false, false><<<ggrid(N_NODES, DFEAT), blk256, 0, stream>>>(
        h1, wt_a2, a_b2, bn2_g, bn2_b, x, x_t, nullptr, N_NODES, DFEAT, HIDF);

    // ---- KAN head ----
    int nd = N_NODES * DFEAT;
    add_k<<<nd / 256, blk256, 0, stream>>>(x_t, x_m, hbuf, nd);

    flatten_spline_t_k<<<cgrid(DFEAT * KSPL), blk256, 0, stream>>>(k1_sp, k1_sc, wt_sp);
    bspline_k<<<nd / 256, blk256, 0, stream>>>(hbuf, bs, nd);
    gemm_wmma<float, true, false, false><<<ggrid(N_NODES, DFEAT), blk256, 0, stream>>>(
        hbuf, wt_k1, nullptr, nullptr, nullptr, nullptr, kmid, nullptr, N_NODES, DFEAT, DFEAT);
    gemm_wmma<__bf16, false, false, true><<<ggrid(N_NODES, DFEAT), blk256, 0, stream>>>(
        bs, wt_sp, nullptr, nullptr, nullptr, nullptr, kmid, nullptr, N_NODES, DFEAT, KSPL);

    flatten_spline_t_k<<<cgrid(DFEAT * KSPL), blk256, 0, stream>>>(k2_sp, k2_sc, wt_sp);
    bspline_k<<<nd / 256, blk256, 0, stream>>>(kmid, bs, nd);
    gemm_wmma<float, true, false, false><<<ggrid(N_NODES, DFEAT), blk256, 0, stream>>>(
        kmid, wt_k2, nullptr, nullptr, nullptr, nullptr, out, nullptr, N_NODES, DFEAT, DFEAT);
    gemm_wmma<__bf16, false, false, true><<<ggrid(N_NODES, DFEAT), blk256, 0, stream>>>(
        bs, wt_sp, nullptr, nullptr, nullptr, nullptr, out, nullptr, N_NODES, DFEAT, KSPL);
}
